// SSUnetDenoiser_3169685864798
// MI455X (gfx1250) — matrix operand order repaired and checked by reading, not compiled, not run on hardware
//
#include <hip/hip_runtime.h>
#include <hip/hip_bf16.h>

typedef __attribute__((ext_vector_type(16))) _Float16 v16h;
typedef __attribute__((ext_vector_type(8)))  float    v8f;

#define WPB 4                 // waves per block (wave32)
#define CONV_BLOCK (WPB * 32)

// ---------------------------------------------------------------------------
// Generic 3x3x3 conv as implicit GEMM on v_wmma_f32_16x16x32_f16.
//   MODE 0: stride 1, pad 1           (SubMConv3d dense equivalent)
//   MODE 1: stride 2, pad 1           (downsample)
//   MODE 2: lhs_dilation 2, pad (1,2) (SparseInverseConv3d / transpose conv)
// CIN/MODE/PRE/RESID are compile-time. Tile coordinates go through
// readfirstlane so plane/row math and most bounds checks are SALU; the gather
// offset select is a single unsigned add under one cndmask.
// ---------------------------------------------------------------------------
template<int CIN, int MODE, bool PRE, bool RESID>
__global__ __launch_bounds__(CONV_BLOCK)
void conv3d_wmma(const float* __restrict__ in, const float* __restrict__ wgt,
                 const float* __restrict__ bias, const float* __restrict__ resid,
                 const float* __restrict__ pre, float* __restrict__ out,
                 int D, int H, int W, int Do, int Ho, int Wo, int Cout)
{
  constexpr int TPI   = 32 / CIN;               // taps per K=32 slab
  constexpr int ITERS = (27 + TPI - 1) / TPI;   // K slabs
  constexpr int KP    = ITERS * 32;
  constexpr int CPB   = (CIN < 8) ? CIN : 8;    // contiguous channels per run
  constexpr int RUNS  = 16 / CPB;               // runs per lane (16 A elements)

  __shared__ __align__(32) _Float16 lw[16 * KP];

  // Stage weights -> LDS as f16 B-fragment layout lw[n*KP + it*32 + 16h + s] holding table-order k (bits 3,4 of k swapped)
  for (int idx = threadIdx.x; idx < 16 * KP; idx += CONV_BLOCK) {
    int n = idx / KP, k = idx % KP;
    int it = k >> 5, kk = (k & 7) | ((k & 8) << 1) | ((k & 16) >> 1);
    int tap = it * TPI + kk / CIN;
    int c   = kk % CIN;
    float v = 0.f;
    if (tap < 27 && n < Cout) v = wgt[(tap * CIN + c) * Cout + n];
    lw[idx] = (_Float16)v;
  }
  __syncthreads();

  const int ntw   = (Wo + 15) >> 4;
  // wave-uniform tile id -> force into SGPRs so dz/hy/w0 math is scalar
  const int tile  = __builtin_amdgcn_readfirstlane(blockIdx.x * WPB + (threadIdx.x >> 5));
  const int tiles = Do * Ho * ntw;
  if (tile >= tiles) return;
  const int wt = tile % ntw;
  const int r0 = tile / ntw;
  const int hy = r0 % Ho;
  const int dz = r0 / Ho;
  const int w0 = wt << 4;

  const int lane = threadIdx.x & 31;
  const int mrow = lane & 15;     // A-fragment row this lane gathers
  const int half = lane >> 4;     // K half selector (CDNA5 16-bit A layout)
  const int kh8  = half * 8;
  const int ow   = w0 + mrow;

  // Hoist BN scale/shift for this lane's (run-invariant) channel base
  float ps[CPB], pb[CPB];
  if constexpr (PRE) {
    const int c0b = (CIN == 16) ? (half << 3) : 0;
#pragma unroll
    for (int c = 0; c < CPB; ++c) {
      ps[c] = pre[c0b + c];
      pb[c] = pre[16 + c0b + c];
    }
  }

  v8f acc = {};
#pragma unroll
  for (int it = 0; it < ITERS; ++it) {
    v16h a;
#pragma unroll
    for (int r = 0; r < RUNS; ++r) {
      const int blk     = (r * CPB) >> 3;            // which 8-wide A block
      const int woff    = r * CPB - 8 * blk;
      const int k_start = 16 * blk + kh8 + woff;     // first k of this run
      const int tap     = it * TPI + k_start / CIN;  // shift (CIN = 2^n)
      const int c0      = k_start % CIN;             // channel base of run

      const int kd  = tap / 9;
      const int rem = tap - kd * 9;
      const int khh = rem / 3;
      const int kw  = rem - khh * 3;

      int id, ih, iwv;           // id/ih scalar-ish (dz,hy uniform), iwv per-lane
      bool par = true;
      if constexpr (MODE == 0) {
        id = dz + kd - 1;  ih = hy + khh - 1;  iwv = ow + kw - 1;
      } else if constexpr (MODE == 1) {
        id = 2*dz + kd - 1;  ih = 2*hy + khh - 1;  iwv = 2*ow + kw - 1;
      } else {  // transpose: out[o] += w[k]*in[(o+k-1)/2] when (o+k-1) even
        const int td = dz + kd - 1, th = hy + khh - 1, tw = ow + kw - 1;
        par = !((td | th | tw) & 1);
        id = td >> 1; ih = th >> 1; iwv = tw >> 1;
      }
      const bool inb = (tap < 27) && par &&
                       (unsigned)id  < (unsigned)D &&
                       (unsigned)ih  < (unsigned)H &&
                       (unsigned)iwv < (unsigned)W;
      // scalar plane/row offset + per-lane column term; single select
      const unsigned rowoff = (unsigned)((id * H + ih) * W) * CIN;
      const unsigned off    = inb ? (rowoff + (unsigned)iwv * CIN + c0) : 0u;
      const float* src = in + off;    // invalid runs read in[0] (always mapped)

      float buf[CPB];
      if constexpr (CPB == 8) {
        const float4 f0 = *(const float4*)src;
        const float4 f1 = *(const float4*)(src + 4);
        buf[0]=f0.x; buf[1]=f0.y; buf[2]=f0.z; buf[3]=f0.w;
        buf[4]=f1.x; buf[5]=f1.y; buf[6]=f1.z; buf[7]=f1.w;
      } else if constexpr (CPB == 4) {
        const float4 f0 = *(const float4*)src;
        buf[0]=f0.x; buf[1]=f0.y; buf[2]=f0.z; buf[3]=f0.w;
      } else {
        buf[0] = *src;
      }

#pragma unroll
      for (int c = 0; c < CPB; ++c) {
        float v = buf[c];
        if constexpr (PRE) {                      // fused BatchNorm + ReLU
          v = v * ps[c] + pb[c];
          v = v > 0.f ? v : 0.f;
        }
        v = inb ? v : 0.f;                        // zero padding (post-transform)
        a[r * CPB + c] = (_Float16)v;
      }
    }
    // B fragment: lane column = lane&15, K group = half*16; 32B-aligned LDS
    v16h b = *(const v16h*)(lw + (lane & 15) * KP + it * 32 + half * 16);
    acc = __builtin_amdgcn_wmma_f32_16x16x32_f16(false, a, false, b,
                                                 (short)0, acc, false, false);
  }

  // Epilogue: CDNA5 f32 C/D layout (VGPR r -> M=r / M=8+r split at lane 16)
  const int ncol   = lane & 15;
  const bool colok = ncol < Cout;
  float biasv = 0.f;
  if (bias && colok) biasv = bias[ncol];
  const int sbase = ((dz * Ho + hy) * Wo + w0) * Cout;   // scalar
  const int mb    = half << 3;
#pragma unroll
  for (int r = 0; r < 8; ++r) {
    if (colok && (w0 + mb + r) < Wo) {
      const unsigned idx = (unsigned)(sbase + (mb + r) * Cout + ncol);
      float v = acc[r] + biasv;
      if constexpr (RESID) v += resid[idx];
      out[idx] = v;
    }
  }
}

// ---------------------------------------------------------------------------
// BatchNorm batch statistics (biased variance) and small helpers
// ---------------------------------------------------------------------------
__global__ void zero_k(float* p, int n) {
  int i = blockIdx.x * blockDim.x + threadIdx.x;
  if (i < n) p[i] = 0.f;
}

__global__ void bn_stats(const float* __restrict__ x, float* __restrict__ st,
                         int nvox, int C) {
  int gtid   = blockIdx.x * blockDim.x + threadIdx.x;
  int total  = gridDim.x * blockDim.x;
  int c      = gtid % C;
  int v0     = gtid / C;
  int stride = total / C;
  float s = 0.f, q = 0.f;
  for (int v = v0; v < nvox; v += stride) {
    float t = x[v * C + c];
    s += t; q += t * t;
  }
  atomicAdd(&st[c], s);
  atomicAdd(&st[16 + c], q);
}

__global__ void bn_finalize(const float* __restrict__ st, const float* __restrict__ g,
                            const float* __restrict__ b, float* __restrict__ ss,
                            float invn, int C) {
  int c = threadIdx.x;
  if (c < C) {
    float mean = st[c] * invn;
    float var  = st[16 + c] * invn - mean * mean;
    float sc   = g[c] * rsqrtf(var + 1e-4f);
    ss[c]      = sc;
    ss[16 + c] = b[c] - mean * sc;
  }
}

__global__ void concat2(const float* __restrict__ a, const float* __restrict__ b,
                        float* __restrict__ o, int V) {
  int i = blockIdx.x * blockDim.x + threadIdx.x;
  if (i < V * 16) {
    int v = i >> 4, c = i & 15;
    o[i] = (c < 8) ? a[v * 8 + c] : b[v * 8 + (c - 8)];
  }
}

__global__ void mask_k(const float* __restrict__ logits, const float* __restrict__ xin,
                       float* __restrict__ mo, int n) {
  int i = blockIdx.x * blockDim.x + threadIdx.x;
  if (i < n) mo[i] = logits[i] > 0.f ? xin[i] : 0.f;
}

// ---------------------------------------------------------------------------
extern "C" void kernel_launch(void* const* d_in, const int* in_sizes, int n_in,
                              void* d_out, int out_size, void* d_ws, size_t ws_size,
                              hipStream_t stream) {
  (void)in_sizes; (void)n_in; (void)out_size; (void)ws_size;

  // ---- unpack parameters in setup_inputs() dict insertion order ----
  int ii = 0;
  auto F = [&]() { return (const float*)d_in[ii++]; };
  const float* x_in = F();  // [1,96,96,96,1]

  const float *h_w0 = F(), *h_g0 = F(), *h_be0 = F();
  const float *h_w1 = F(), *h_g1 = F(), *h_be1 = F();
  const float *h_w2 = F();

  struct RB  { const float *g1,*b1,*w1,*bb1,*g2,*b2,*w2,*bb2,*wi,*bi; };
  struct SMP { const float *g,*b,*w,*bb; };
  auto loadRB = [&](bool shortcut) {
    RB r; r.g1=F(); r.b1=F(); r.w1=F(); r.bb1=F();
    r.g2=F(); r.b2=F(); r.w2=F(); r.bb2=F();
    if (shortcut) { r.wi=F(); r.bi=F(); } else { r.wi=nullptr; r.bi=nullptr; }
    return r;
  };
  auto loadS = [&]() { SMP s; s.g=F(); s.b=F(); s.w=F(); s.bb=F(); return s; };

  RB  p00 = loadRB(false); SMP ds0 = loadS();
  RB  p10 = loadRB(false); SMP ds1 = loadS();
  RB  p20 = loadRB(false); SMP ds2 = loadS();
  RB  p3a = loadRB(false), p3b = loadRB(false);
  SMP up2 = loadS(); RB p21  = loadRB(true);
  SMP up1 = loadS(); RB p11  = loadRB(true);
  SMP up0 = loadS(); RB p01a = loadRB(true); RB p01b = loadRB(false);
  const float *t_g0=F(), *t_be0=F(), *t_w0=F();
  const float *t_g1=F(), *t_be1=F(), *t_w1=F();
  const float *t_g2=F(), *t_be2=F(), *t_w2=F(), *t_b2=F();

  // ---- workspace carve ----
  const int V0 = 96*96*96, V1 = 48*48*48, V2 = 24*24*24, V3 = 12*12*12;
  float* p = (float*)d_ws;
  auto alloc = [&](long n) { float* q = p; p += n; return q; };
  float* stats = alloc(32);
  float* scsh  = alloc(32);
  float *T0a=alloc((long)V0*8), *T0b=alloc((long)V0*8), *T0c=alloc((long)V0*8);
  float *S0 =alloc((long)V0*8), *X0 =alloc((long)V0*16);
  float *T1a=alloc((long)V1*8), *T1b=alloc((long)V1*8), *T1c=alloc((long)V1*8);
  float *S1 =alloc((long)V1*8), *X1 =alloc((long)V1*16);
  float *T2a=alloc((long)V2*8), *T2b=alloc((long)V2*8), *T2c=alloc((long)V2*8);
  float *S2 =alloc((long)V2*8), *X2 =alloc((long)V2*16);
  float *T3a=alloc((long)V3*8), *T3b=alloc((long)V3*8), *T3c=alloc((long)V3*8);

  // ---- launch helpers ----
  auto bn = [&](const float* x, const float* g, const float* b, int nvox, int C) {
    zero_k<<<1, 32, 0, stream>>>(stats, 32);
    bn_stats<<<512, 256, 0, stream>>>(x, stats, nvox, C);
    bn_finalize<<<1, 32, 0, stream>>>(stats, g, b, scsh, 1.0f / (float)nvox, C);
  };
  auto conv = [&](const float* in, const float* w, const float* bias, const float* resid,
                  bool pre, float* out, int D, int H, int W,
                  int Do, int Ho, int Wo, int Ci, int Co, int mode) {
    int ntw    = (Wo + 15) / 16;
    int tiles  = Do * Ho * ntw;
    int blocks = (tiles + WPB - 1) / WPB;
    dim3 g(blocks), blk(CONV_BLOCK);
    if (mode == 1) {
      conv3d_wmma<8,1,true,false><<<g,blk,0,stream>>>(in,w,bias,resid,scsh,out,D,H,W,Do,Ho,Wo,Co);
    } else if (mode == 2) {
      conv3d_wmma<8,2,true,false><<<g,blk,0,stream>>>(in,w,bias,resid,scsh,out,D,H,W,Do,Ho,Wo,Co);
    } else if (Ci == 1) {
      conv3d_wmma<1,0,false,false><<<g,blk,0,stream>>>(in,w,bias,resid,scsh,out,D,H,W,Do,Ho,Wo,Co);
    } else if (Ci == 4) {
      conv3d_wmma<4,0,true,false><<<g,blk,0,stream>>>(in,w,bias,resid,scsh,out,D,H,W,Do,Ho,Wo,Co);
    } else if (Ci == 8) {
      if (resid)
        conv3d_wmma<8,0,true,true><<<g,blk,0,stream>>>(in,w,bias,resid,scsh,out,D,H,W,Do,Ho,Wo,Co);
      else
        conv3d_wmma<8,0,true,false><<<g,blk,0,stream>>>(in,w,bias,resid,scsh,out,D,H,W,Do,Ho,Wo,Co);
    } else {  // Ci == 16
      if (pre)
        conv3d_wmma<16,0,true,false><<<g,blk,0,stream>>>(in,w,bias,resid,scsh,out,D,H,W,Do,Ho,Wo,Co);
      else  // shortcut conv: no BN/ReLU, adds the residual branch
        conv3d_wmma<16,0,false,true><<<g,blk,0,stream>>>(in,w,bias,resid,scsh,out,D,H,W,Do,Ho,Wo,Co);
    }
  };
  auto resblock = [&](const float* x, const RB& rb, float* out, float* t1, float* t2,
                      int Dd, int Ci) {
    int nv = Dd * Dd * Dd;
    bn(x, rb.g1, rb.b1, nv, Ci);
    conv(x, rb.w1, rb.bb1, nullptr, true, t1, Dd,Dd,Dd, Dd,Dd,Dd, Ci, 8, 0);
    bn(t1, rb.g2, rb.b2, nv, 8);
    if (!rb.wi) {
      conv(t1, rb.w2, rb.bb2, x, true, out, Dd,Dd,Dd, Dd,Dd,Dd, 8, 8, 0);
    } else {
      conv(t1, rb.w2, rb.bb2, nullptr, true, t2, Dd,Dd,Dd, Dd,Dd,Dd, 8, 8, 0);
      conv(x, rb.wi, rb.bi, t2, false, out, Dd,Dd,Dd, Dd,Dd,Dd, Ci, 8, 0);
    }
  };
  auto down = [&](const float* x, const SMP& s, float* out, int Dd) {
    bn(x, s.g, s.b, Dd*Dd*Dd, 8);
    conv(x, s.w, s.bb, nullptr, true, out, Dd,Dd,Dd, Dd/2,Dd/2,Dd/2, 8, 8, 1);
  };
  auto up = [&](const float* x, const SMP& s, float* out, int Dd) {
    bn(x, s.g, s.b, Dd*Dd*Dd, 8);
    conv(x, s.w, s.bb, nullptr, true, out, Dd,Dd,Dd, 2*Dd,2*Dd,2*Dd, 8, 8, 2);
  };
  auto concat = [&](const float* a, const float* b, float* o, int V) {
    int n = V * 16;
    concat2<<<(n + 255) / 256, 256, 0, stream>>>(a, b, o, V);
  };

  // ---- forward pass ----
  // head
  conv(x_in, h_w0, nullptr, nullptr, false, T0a, 96,96,96, 96,96,96, 1, 4, 0);
  bn(T0a, h_g0, h_be0, V0, 4);
  conv(T0a, h_w1, nullptr, nullptr, true, T0b, 96,96,96, 96,96,96, 4, 8, 0);
  bn(T0b, h_g1, h_be1, V0, 8);
  conv(T0b, h_w2, nullptr, nullptr, true, T0a, 96,96,96, 96,96,96, 8, 8, 0);

  // encoder
  resblock(T0a, p00, S0, T0b, nullptr, 96, 8);     // out0
  down(S0, ds0, T1a, 96);                          // 96 -> 48
  resblock(T1a, p10, S1, T1b, nullptr, 48, 8);     // out1
  down(S1, ds1, T2a, 48);                          // 48 -> 24
  resblock(T2a, p20, S2, T2b, nullptr, 24, 8);     // out2
  down(S2, ds2, T3a, 24);                          // 24 -> 12
  resblock(T3a, p3a, T3b, T3c, nullptr, 12, 8);
  resblock(T3b, p3b, T3a, T3c, nullptr, 12, 8);    // out3 in T3a

  // decoder
  up(T3a, up2, T2b, 12);                           // u2 @24^3
  concat(S2, T2b, X2, V2);
  resblock(X2, p21, T2a, T2c, T2b, 24, 16);        // out2' in T2a
  up(T2a, up1, T1b, 24);                           // u1 @48^3
  concat(S1, T1b, X1, V1);
  resblock(X1, p11, T1a, T1c, T1b, 48, 16);        // out1' in T1a
  up(T1a, up0, T0b, 48);                           // u0 @96^3
  concat(S0, T0b, X0, V0);
  resblock(X0, p01a, T0a, T0c, T0b, 96, 16);
  resblock(T0a, p01b, T0c, T0b, nullptr, 96, 8);

  // tail
  bn(T0c, t_g0, t_be0, V0, 8);
  conv(T0c, t_w0, nullptr, nullptr, true, T0a, 96,96,96, 96,96,96, 8, 8, 0);
  bn(T0a, t_g1, t_be1, V0, 8);
  conv(T0a, t_w1, nullptr, nullptr, true, T0b, 96,96,96, 96,96,96, 8, 4, 0);
  bn(T0b, t_g2, t_be2, V0, 4);
  float* outp = (float*)d_out;
  conv(T0b, t_w2, t_b2, nullptr, true, outp, 96,96,96, 96,96,96, 4, 1, 0);  // logits

  // masked = where(logits > 0, inputs, 0)
  mask_k<<<(V0 + 255) / 256, 256, 0, stream>>>(outp, x_in, outp + V0, V0);
}
